// Head_81252191305756
// MI455X (gfx1250) — matrix operand order repaired and checked by reading, not compiled, not run on hardware
//
#include <hip/hip_runtime.h>

typedef __attribute__((ext_vector_type(16))) __bf16 v16bf;
typedef __attribute__((ext_vector_type(8)))  float  v8f;

#define BATCH    8
#define SEQ      2048
#define DDIM     256
#define MTILE    64
#define QTILE    64
#define KTILE    64
#define NTHREADS 128

#if __has_builtin(__builtin_amdgcn_tensor_load_to_lds) && \
    __has_builtin(__builtin_amdgcn_s_wait_tensorcnt)
#define USE_TDM 1
#else
#define USE_TDM 0
#endif

// native f32 -> bf16 (RNE) via hardware convert
static __device__ __forceinline__ unsigned short f2bf(float f) {
    union { __bf16 h; unsigned short s; } u;
    u.h = (__bf16)f;
    return u.s;
}

union FragAB { v16bf v; unsigned u[8]; };

// A-fragment 16x32 bf16 from row-major src (ld elements/row), tile at (row0, k0).
// lanes 0-15: M=lane, K={k0..k0+15}; lanes 16-31: M=lane-16, K={k0+16..k0+31}
static __device__ __forceinline__ v16bf load_a_frag(const unsigned short* base, int ld,
                                                    int row0, int k0) {
    const int lane = threadIdx.x & 31;
    const int half = lane >> 4;
    const int m    = lane & 15;
    const unsigned short* p = base + (size_t)(row0 + m) * ld + k0 + half * 16;
    FragAB f;
#pragma unroll
    for (int j = 0; j < 4; ++j) f.u[j]     = *(const unsigned*)(p + 2 * j);
#pragma unroll
    for (int j = 0; j < 4; ++j) f.u[4 + j] = *(const unsigned*)(p + 8 + 2 * j);
    return f.v;
}

// B-fragment 32x16 bf16 where storage is "n-major": src[n*ld + k] (K contiguous).
static __device__ __forceinline__ v16bf load_b_frag(const unsigned short* base, int ld,
                                                    int k0, int n0) {
    const int lane = threadIdx.x & 31;
    const int n    = n0 + (lane & 15);
    const int koff = k0 + (lane >> 4) * 16;
    const unsigned short* p = base + (size_t)n * ld + koff;
    FragAB f;
#pragma unroll
    for (int j = 0; j < 8; ++j) f.u[j] = *(const unsigned*)(p + 2 * j);
    return f.v;
}

static __device__ __forceinline__ v8f wmma_bf16(v16bf a, v16bf b, v8f c) {
    return __builtin_amdgcn_wmma_f32_16x16x32_bf16(false, a, false, b, (short)0, c,
                                                   false, false);
}

#if USE_TDM
typedef unsigned int u32x4 __attribute__((ext_vector_type(4)));
typedef int          i32x4 __attribute__((ext_vector_type(4)));
typedef int          i32x8 __attribute__((ext_vector_type(8)));

// Issue a TDM 2D tile load (bf16 elements): tile_d0 x tile_d1 tile from a
// row-major tensor with row length tensor_d0 (elements) and row stride
// stride0 (elements) into contiguous LDS at lds_off.
static __device__ __forceinline__ void tdm_load_2d(unsigned lds_off,
                                                   const void* gptr,
                                                   unsigned tensor_d0,
                                                   unsigned tensor_d1,
                                                   unsigned long long stride0,
                                                   unsigned tile_d0,
                                                   unsigned tile_d1) {
    const unsigned long long ga = (unsigned long long)(size_t)gptr;
    // D# group 0: count=1 | lds_addr | global_addr[56:0] | type=2 (bits 127:126)
    u32x4 g0;
    g0[0] = 1u;                                      // count=1, user mode
    g0[1] = lds_off;                                 // lds_addr
    g0[2] = (unsigned)(ga & 0xFFFFFFFFu);            // global_addr[31:0]
    g0[3] = (unsigned)((ga >> 32) & 0x01FFFFFFu) | (2u << 30);  // addr[56:32] | type=2
    // D# group 1
    i32x8 g1;
    g1[0] = (int)(1u << 16);                         // wg_mask=0, data_size=1 (2 bytes)
    g1[1] = (int)((tensor_d0 & 0xFFFFu) << 16);      // barrier_addr=0 | tensor_dim0[15:0]
    g1[2] = (int)(((tensor_d0 >> 16) & 0xFFFFu) | ((tensor_d1 & 0xFFFFu) << 16));
    g1[3] = (int)(((tensor_d1 >> 16) & 0xFFFFu) | (tile_d0 << 16));   // | tile_dim0
    g1[4] = (int)(tile_d1 & 0xFFFFu);                // tile_dim1 | tile_dim2=0
    g1[5] = (int)(stride0 & 0xFFFFFFFFu);            // tensor_dim0_stride[31:0]
    g1[6] = (int)((stride0 >> 32) & 0xFFFFu);        // stride[47:32] | dim1_stride lo = 0
    g1[7] = 0;
    i32x4 z4 = {0, 0, 0, 0};
#if defined(__clang_major__) && __clang_major__ >= 23
    i32x8 z8 = {0, 0, 0, 0, 0, 0, 0, 0};
    __builtin_amdgcn_tensor_load_to_lds(g0, g1, z4, z4, z8, 0);
#else
    __builtin_amdgcn_tensor_load_to_lds(g0, g1, z4, z4, 0);
#endif
}
#endif  // USE_TDM

// ---------------------------------------------------------------------------
// Kernel 1: q = x Wq^T, k = x Wk^T, v = x Wv^T  (bf16 outputs; v stored transposed)
// ---------------------------------------------------------------------------
__global__ __launch_bounds__(NTHREADS) void qkv_proj_kernel(
    const float* __restrict__ x,  const float* __restrict__ Wq,
    const float* __restrict__ Wk, const float* __restrict__ Wv,
    unsigned short* __restrict__ qout, unsigned short* __restrict__ kout,
    unsigned short* __restrict__ vtout)
{
    __shared__ __align__(16) unsigned short Xt[MTILE * DDIM];  // 32 KB
    __shared__ __align__(16) unsigned short Wt[64 * DDIM];     // 32 KB

    const int t      = threadIdx.x;
    const int wave   = t >> 5;
    const int lane   = t & 31;
    const int half   = lane >> 4;
    const int col    = lane & 15;
    const int row0   = wave * 16;
    const int mblock = blockIdx.x * MTILE;

    { // stage x tile -> bf16 LDS
        const float4* src = (const float4*)(x + (size_t)mblock * DDIM);
#pragma unroll
        for (int i = 0; i < (MTILE * DDIM / 4) / NTHREADS; ++i) {
            int idx = t + i * NTHREADS;
            float4 v4 = src[idx];
            int o = idx * 4;
            Xt[o + 0] = f2bf(v4.x); Xt[o + 1] = f2bf(v4.y);
            Xt[o + 2] = f2bf(v4.z); Xt[o + 3] = f2bf(v4.w);
        }
    }
    __syncthreads();

    v16bf afrag[8];
#pragma unroll
    for (int ks = 0; ks < 8; ++ks) afrag[ks] = load_a_frag(Xt, DDIM, row0, ks * 32);

    const float* Ws[3] = {Wq, Wk, Wv};
#pragma unroll
    for (int mat = 0; mat < 3; ++mat) {
        const float* W = Ws[mat];
        for (int et = 0; et < 4; ++et) {           // 4 e-tiles of 64 output cols
            __syncthreads();
            const float4* wsrc = (const float4*)(W + (size_t)(et * 64) * DDIM);
#pragma unroll
            for (int i = 0; i < (64 * DDIM / 4) / NTHREADS; ++i) {
                int idx = t + i * NTHREADS;
                float4 v4 = wsrc[idx];
                int o = idx * 4;
                Wt[o + 0] = f2bf(v4.x); Wt[o + 1] = f2bf(v4.y);
                Wt[o + 2] = f2bf(v4.z); Wt[o + 3] = f2bf(v4.w);
            }
            __syncthreads();
#pragma unroll
            for (int nt = 0; nt < 4; ++nt) {
                v8f acc = {};
                v16bf bcur = load_b_frag(Wt, DDIM, 0, nt * 16);
#pragma unroll
                for (int ks = 0; ks < 8; ++ks) {   // double-buffered B fragments
                    v16bf bnext = (ks < 7) ? load_b_frag(Wt, DDIM, (ks + 1) * 32, nt * 16)
                                           : bcur;
                    acc = wmma_bf16(afrag[ks], bcur, acc);
                    bcur = bnext;
                }
                const int e = et * 64 + nt * 16 + col;
#pragma unroll
                for (int r = 0; r < 8; ++r) {
                    const int grow = mblock + row0 + r + half * 8;
                    const unsigned short bv = f2bf(acc[r]);
                    if (mat == 0)      qout[(size_t)grow * DDIM + e] = bv;
                    else if (mat == 1) kout[(size_t)grow * DDIM + e] = bv;
                    else {
                        const int bb = grow >> 11;            // batch
                        const int nn = grow & (SEQ - 1);      // seq pos
                        vtout[((size_t)bb * DDIM + e) * SEQ + nn] = bv;
                    }
                }
            }
        }
    }
}

// ---------------------------------------------------------------------------
// Kernel 2: fused flash attention, out = (softmax(q k^T) * D^-1/2) @ v
// ---------------------------------------------------------------------------
__global__ __launch_bounds__(NTHREADS) void flash_attn_kernel(
    const unsigned short* __restrict__ q, const unsigned short* __restrict__ k,
    const unsigned short* __restrict__ vt, float* __restrict__ out)
{
    __shared__ __align__(16) unsigned short Kt[KTILE * DDIM];      // 32 KB, [kv][d]
    __shared__ __align__(16) unsigned short Vt[DDIM * KTILE];      // 32 KB, [d][kv]
    __shared__ __align__(16) unsigned short Pt[4 * 16 * KTILE];    //  8 KB, per-wave P

    const int t    = threadIdx.x;
    const int wave = t >> 5;
    const int lane = t & 31;
    const int half = lane >> 4;
    const int col  = lane & 15;

    const int b  = blockIdx.x >> 5;                 // 32 q-tiles per batch
    const int qt = blockIdx.x & 31;
    const size_t qkbase = (size_t)b * SEQ * DDIM;
    const int qrow0 = qt * QTILE + wave * 16;

    // q fragments: 16 rows x 256 (kept in VGPRs for the whole kernel)
    v16bf qfrag[8];
#pragma unroll
    for (int ks = 0; ks < 8; ++ks)
        qfrag[ks] = load_a_frag(q + qkbase, DDIM, qrow0, ks * 32);

    v8f oacc[16];
    v8f zero = {};
#pragma unroll
    for (int i = 0; i < 16; ++i) oacc[i] = zero;
    float mrun[8], lrun[8];
#pragma unroll
    for (int r = 0; r < 8; ++r) { mrun[r] = -3.0e38f; lrun[r] = 0.0f; }

    for (int kt = 0; kt < SEQ / KTILE; ++kt) {
        __syncthreads();
#if USE_TDM
        if (wave == 0) {
            // K tile: rows kt*64..+63 of [2048][256] bf16, contiguous -> Kt
            tdm_load_2d((unsigned)(size_t)Kt,
                        k + qkbase + (size_t)kt * KTILE * DDIM,
                        DDIM, SEQ, DDIM, DDIM, KTILE);
            // vT tile: 256 rows x 64 cols of [256][2048] bf16, stride 2048 -> Vt
            tdm_load_2d((unsigned)(size_t)Vt,
                        vt + (size_t)b * DDIM * SEQ + (size_t)kt * KTILE,
                        SEQ, DDIM, SEQ, KTILE, DDIM);
            __builtin_amdgcn_s_wait_tensorcnt(0);
        }
#else
        { // K tile -> LDS (bf16, b128 copies)
            const uint4* src = (const uint4*)(k + qkbase + (size_t)kt * KTILE * DDIM);
            uint4* dst = (uint4*)Kt;
#pragma unroll
            for (int i = 0; i < (KTILE * DDIM / 8) / NTHREADS; ++i)
                dst[t + i * NTHREADS] = src[t + i * NTHREADS];
        }
        { // vT tile -> LDS: rows d=0..255, cols kt*64..+63
            const unsigned short* vbase = vt + (size_t)b * DDIM * SEQ + kt * KTILE;
            uint4* dst = (uint4*)Vt;
#pragma unroll
            for (int i = 0; i < (DDIM * KTILE / 8) / NTHREADS; ++i) {
                int idx = t + i * NTHREADS;
                int d = idx >> 3;
                int c = idx & 7;
                dst[idx] = *(const uint4*)(vbase + (size_t)d * SEQ + c * 8);
            }
        }
        if (kt + 1 < SEQ / KTILE)   // prefetch next K tile (global_prefetch_b8)
            __builtin_prefetch(k + qkbase + (size_t)(kt + 1) * KTILE * DDIM + t * 32);
#endif
        __syncthreads();

        // S = q_tile @ k_tile^T (K stored [ki][d]: n-major, K contiguous)
        v8f sacc[4];
#pragma unroll
        for (int nt = 0; nt < 4; ++nt) {
            v8f acc = {};
            v16bf bcur = load_b_frag(Kt, DDIM, 0, nt * 16);
#pragma unroll
            for (int ks = 0; ks < 8; ++ks) {       // double-buffered B fragments
                v16bf bnext = (ks < 7) ? load_b_frag(Kt, DDIM, (ks + 1) * 32, nt * 16)
                                       : bcur;
                acc = wmma_bf16(qfrag[ks], bcur, acc);
                bcur = bnext;
            }
            sacc[nt] = acc;
        }

        // Online softmax. Row r lives in lanes 0-15 (rows r) / 16-31 (rows 8+r);
        // xor-shuffles with masks 1..8 reduce within each 16-lane half.
        unsigned short* Pw = Pt + wave * 16 * KTILE;
#pragma unroll
        for (int r = 0; r < 8; ++r) {
            float cand = fmaxf(fmaxf(sacc[0][r], sacc[1][r]),
                               fmaxf(sacc[2][r], sacc[3][r]));
#pragma unroll
            for (int mask = 1; mask < 16; mask <<= 1)
                cand = fmaxf(cand, __shfl_xor(cand, mask, 32));
            const float mnew = fmaxf(mrun[r], cand);
            const float corr = __expf(mrun[r] - mnew);
            float rs = 0.0f;
#pragma unroll
            for (int nt = 0; nt < 4; ++nt) {
                float p = __expf(sacc[nt][r] - mnew);
                rs += p;
                Pw[(r + half * 8) * KTILE + nt * 16 + col] = f2bf(p);
            }
#pragma unroll
            for (int mask = 1; mask < 16; mask <<= 1)
                rs += __shfl_xor(rs, mask, 32);
            lrun[r] = lrun[r] * corr + rs;
            mrun[r] = mnew;
#pragma unroll
            for (int nt = 0; nt < 16; ++nt) oacc[nt][r] *= corr;
        }
        __builtin_amdgcn_wave_barrier();
        asm volatile("s_wait_dscnt 0" ::: "memory");   // cross-lane LDS RAW inside wave

        // O += P @ V : Vt is [d][kv] so B pairs (kv,kv+1) are contiguous dwords.
        v16bf pf0 = load_a_frag(Pw, KTILE, 0, 0);
        v16bf pf1 = load_a_frag(Pw, KTILE, 0, 32);
        v16bf b0 = load_b_frag(Vt, KTILE, 0,  0);
        v16bf b1 = load_b_frag(Vt, KTILE, 32, 0);
#pragma unroll
        for (int nt = 0; nt < 16; ++nt) {          // double-buffered B fragments
            v16bf bn0 = (nt < 15) ? load_b_frag(Vt, KTILE, 0,  (nt + 1) * 16) : b0;
            v16bf bn1 = (nt < 15) ? load_b_frag(Vt, KTILE, 32, (nt + 1) * 16) : b1;
            v8f acc = oacc[nt];
            acc = wmma_bf16(pf0, b0, acc);
            acc = wmma_bf16(pf1, b1, acc);
            oacc[nt] = acc;
            b0 = bn0; b1 = bn1;
        }
    }

    // epilogue: reference scales AFTER softmax -> out = (O / l) * D^-1/2
    const float scaling = 0.0625f;   // 256^-0.5
#pragma unroll
    for (int r = 0; r < 8; ++r) {
        const float inv = scaling / lrun[r];
        const int grow = b * SEQ + qt * QTILE + wave * 16 + r + half * 8;
#pragma unroll
        for (int nt = 0; nt < 16; ++nt)
            out[(size_t)grow * DDIM + nt * 16 + col] = oacc[nt][r] * inv;
    }
}

extern "C" void kernel_launch(void* const* d_in, const int* in_sizes, int n_in,
                              void* d_out, int out_size, void* d_ws, size_t ws_size,
                              hipStream_t stream) {
    (void)in_sizes; (void)n_in; (void)out_size; (void)ws_size;
    const float* x  = (const float*)d_in[0];
    const float* Wq = (const float*)d_in[1];
    const float* Wk = (const float*)d_in[2];
    const float* Wv = (const float*)d_in[3];

    // workspace: q, k (row-major bf16) and v transposed (bf16), 25.2 MB total
    unsigned short* qb  = (unsigned short*)d_ws;
    unsigned short* kb  = qb + (size_t)BATCH * SEQ * DDIM;
    unsigned short* vtb = kb + (size_t)BATCH * SEQ * DDIM;
    float* out = (float*)d_out;

    qkv_proj_kernel<<<(BATCH * SEQ) / MTILE, NTHREADS, 0, stream>>>(
        x, Wq, Wk, Wv, qb, kb, vtb);
    flash_attn_kernel<<<BATCH * (SEQ / QTILE), NTHREADS, 0, stream>>>(
        qb, kb, vtb, out);
}
